// rootfind_module_57243324121136
// MI455X (gfx1250) — matrix operand order repaired and checked by reading, not compiled, not run on hardware
//
#include <hip/hip_runtime.h>
#include <hip/hip_bf16.h>
#include <stdint.h>

// Problem constants (match reference)
#define BB     8192
#define NND    256
#define HHD    1024
#define BETA_C 0.99f
#define TOL_C  1e-4f
#define MAXIT  100

typedef __attribute__((ext_vector_type(16))) __bf16 v16bf;
typedef __attribute__((ext_vector_type(8)))  __bf16 v8bf;
typedef __attribute__((ext_vector_type(8)))  float  v8f;
typedef __attribute__((ext_vector_type(4)))  unsigned int v4u;
typedef __attribute__((ext_vector_type(8)))  int   v8i;
typedef __attribute__((ext_vector_type(4)))  int   v4i;

union Frag { v16bf v; v8bf h[2]; };

// LDS row stride in bf16 elements: 32 data + 4-DWORD pad = 80 bytes.
// 16B-aligned for v8bf reads; 80B = 20 banks -> conflict-free b128 frag reads.
#define LDK 40

// ---------------------------------------------------------------------------
// TDM availability (probe-verified builtins; arity differs per toolchain)
// ---------------------------------------------------------------------------
#if defined(__has_builtin)
# if __has_builtin(__builtin_amdgcn_tensor_load_to_lds) && \
     __has_builtin(__builtin_amdgcn_s_wait_tensorcnt)
#  define USE_TDM 1
# endif
#endif
#ifndef USE_TDM
# define USE_TDM 0
#endif

#if USE_TDM
// Build a 2-D Tensor-DMA descriptor (CDNA5 ISA) and
// issue TENSOR_LOAD_TO_LDS: tile = tileRows x 32 bf16, row stride = tensorK
// elements in global, packed into LDS with 16B pad per 64B row (stride LDK).
__device__ __forceinline__ void tdm_issue(unsigned long long ga, unsigned ldsOff,
                                          unsigned tensorK, unsigned tensorRows,
                                          unsigned tileRows)
{
  v4u g0;
  g0[0] = 1u;                                         // count=1 (valid), user mode
  g0[1] = ldsOff;                                     // lds_addr
  g0[2] = (unsigned)(ga & 0xFFFFFFFFu);               // global_addr[31:0]
  g0[3] = (unsigned)((ga >> 32) & 0x01FFFFFFu)        // global_addr[56:32]
        | (2u << 30);                                 // type=2 ("image")

  v8i g1;
  g1[0] = (int)((1u << 16)      // data_size = 1 -> 2 bytes (bf16)
              | (1u << 20)      // pad_enable
              | (3u << 22)      // pad_interval = 3 -> every 16 DWORDs (64B row)
              | (3u << 25));    // pad_amount   = 3 -> 4 DWORDs (16B) pad
  g1[1] = (int)((tensorK & 0xFFFFu) << 16);                           // tensor_dim0[15:0]
  g1[2] = (int)(((tensorK >> 16) & 0xFFFFu) |
                ((tensorRows & 0xFFFFu) << 16));                      // dim0 hi | dim1 lo
  g1[3] = (int)(((tensorRows >> 16) & 0xFFFFu) | (32u << 16));        // dim1 hi | tile_dim0=32
  g1[4] = (int)(tileRows & 0xFFFFu);                                  // tile_dim1, tile_dim2=0
  g1[5] = (int)tensorK;                                               // tensor_dim0_stride[31:0]
  g1[6] = 0;                                                          // stride hi | dim1_stride lo
  g1[7] = 0;

  v4i z4 = (v4i)0;
#if defined(__clang_major__) && (__clang_major__ >= 23)
  v8i z8 = (v8i)0;
  __builtin_amdgcn_tensor_load_to_lds(g0, g1, z4, z4, z8, 0);
#else
  __builtin_amdgcn_tensor_load_to_lds(g0, g1, z4, z4, 0);
#endif
}
#endif // USE_TDM

// ---------------------------------------------------------------------------
// Prep: fp32 -> bf16 hi/lo split (elementwise)
// ---------------------------------------------------------------------------
__global__ void split_kernel(const float* __restrict__ in,
                             __bf16* __restrict__ hi, __bf16* __restrict__ lo, int n)
{
  int i = blockIdx.x * blockDim.x + threadIdx.x;
  if (i >= n) return;
  float w = in[i];
  __bf16 h = (__bf16)w;
  hi[i] = h;
  lo[i] = (__bf16)(w - (float)h);
}

// Prep: W (KxN fp32, row-major) -> Wt hi/lo (NxK bf16) via LDS tile transpose
__global__ __launch_bounds__(256)
void transpose_split_kernel(const float* __restrict__ W,
                            __bf16* __restrict__ thi, __bf16* __restrict__ tlo,
                            int K, int N)
{
  __shared__ float t[32][33];
  const int n0 = blockIdx.x * 32, k0 = blockIdx.y * 32;
  const int tx = threadIdx.x & 31, ty = threadIdx.x >> 5;   // ty: 0..7
  #pragma unroll
  for (int j = 0; j < 32; j += 8)
    t[ty + j][tx] = W[(size_t)(k0 + ty + j) * N + n0 + tx];
  __syncthreads();
  #pragma unroll
  for (int j = 0; j < 32; j += 8) {
    float w = t[tx][ty + j];                                 // = W[k0+tx][n0+ty+j]
    __bf16 h = (__bf16)w;
    thi[(size_t)(n0 + ty + j) * K + k0 + tx] = h;
    tlo[(size_t)(n0 + ty + j) * K + k0 + tx] = (__bf16)(w - (float)h);
  }
}

// ---------------------------------------------------------------------------
// bf16x3 WMMA GEMM: C(MxN) = A(MxK) @ B(KxN); A given as hi/lo bf16 row-major,
// B given pre-transposed as Bt hi/lo (NxK bf16 row-major).
// Block: 256 threads = 8 waves (2 M x 4 N); wave tile 32x32 (4 accumulators);
// block tile 64(M) x 128(N); K-step 32; TDM double-buffered LDS staging.
// EPI 0: t=tanh(acc+bias) -> outHi/outLo (h split)
// EPI 1: f=resid+acc+bias -> outF (fp32) + outHi/outLo (fhat split)
// EPI 2: Vout[row] += sum_col acc^2 (atomic)
// ---------------------------------------------------------------------------
template<int EPI>
__global__ __launch_bounds__(256)
void gemm_bf16x3(const __bf16* __restrict__ Ahi, const __bf16* __restrict__ Alo,
                 const __bf16* __restrict__ Bthi, const __bf16* __restrict__ Btlo,
                 const float* __restrict__ bias, const float* __restrict__ resid,
                 float* __restrict__ outF, __bf16* __restrict__ outHi,
                 __bf16* __restrict__ outLo, float* __restrict__ Vout,
                 int M, int N, int K)
{
#if USE_TDM
  __shared__ __align__(16) __bf16 sAhi[2][64][LDK];
  __shared__ __align__(16) __bf16 sAlo[2][64][LDK];
  __shared__ __align__(16) __bf16 sBhi[2][128][LDK];
  __shared__ __align__(16) __bf16 sBlo[2][128][LDK];
#else
  __shared__ __align__(16) __bf16 sAhi[1][64][LDK];
  __shared__ __align__(16) __bf16 sAlo[1][64][LDK];
  __shared__ __align__(16) __bf16 sBhi[1][128][LDK];
  __shared__ __align__(16) __bf16 sBlo[1][128][LDK];
#endif

  const int tid   = threadIdx.x;
  const int lane  = tid & 31;
  const int hl    = lane >> 4;      // ISA fragment half-select
  const int ln    = lane & 15;
  // Wave id forced into an SGPR so ownership branches are scalar (s_cbranch),
  // not exec-mask cascades.
  const int wv    = __builtin_amdgcn_readfirstlane(tid >> 5);   // 0..7
  const int waveM = wv >> 2;        // 0..1  (32-row wave tile)
  const int waveN = wv & 3;         // 0..3  (32-col wave tile)

  const int rowBase = blockIdx.y * 64;
  const int colBase = blockIdx.x * 128;
  const int KT = K >> 5;

  v8f acc[2][2] = {};

  const int abase = hl * 8;         // A frag: K = {abase..+7} U {16+abase..+7}
  const int koff  = hl * 8;         // B frag: K = {koff..+7} U {16+koff..+7}

  // Per-k-step compute on LDS buffer `buf`
  auto compute_tile = [&](int buf) {
    Frag ah[2], al[2], bh[2], bl[2];
    #pragma unroll
    for (int mi = 0; mi < 2; ++mi) {
      const __bf16* ph = &sAhi[buf][waveM * 32 + mi * 16 + ln][0];
      const __bf16* pl = &sAlo[buf][waveM * 32 + mi * 16 + ln][0];
      ah[mi].h[0] = *(const v8bf*)(ph + abase);
      ah[mi].h[1] = *(const v8bf*)(ph + 16 + abase);
      al[mi].h[0] = *(const v8bf*)(pl + abase);
      al[mi].h[1] = *(const v8bf*)(pl + 16 + abase);
    }
    #pragma unroll
    for (int ni = 0; ni < 2; ++ni) {
      const __bf16* ph = &sBhi[buf][waveN * 32 + ni * 16 + ln][0];
      const __bf16* pl = &sBlo[buf][waveN * 32 + ni * 16 + ln][0];
      bh[ni].h[0] = *(const v8bf*)(ph + koff);
      bh[ni].h[1] = *(const v8bf*)(ph + koff + 16);
      bl[ni].h[0] = *(const v8bf*)(pl + koff);
      bl[ni].h[1] = *(const v8bf*)(pl + koff + 16);
    }
    #pragma unroll
    for (int mi = 0; mi < 2; ++mi)
      #pragma unroll
      for (int ni = 0; ni < 2; ++ni) {
        acc[mi][ni] = __builtin_amdgcn_wmma_f32_16x16x32_bf16(
            false, ah[mi].v, false, bh[ni].v, (short)0, acc[mi][ni], false, false);
        acc[mi][ni] = __builtin_amdgcn_wmma_f32_16x16x32_bf16(
            false, ah[mi].v, false, bl[ni].v, (short)0, acc[mi][ni], false, false);
        acc[mi][ni] = __builtin_amdgcn_wmma_f32_16x16x32_bf16(
            false, al[mi].v, false, bh[ni].v, (short)0, acc[mi][ni], false, false);
      }
  };

#if USE_TDM
  // Waves 0..3 each own one tile descriptor; double-buffered pipeline.
  // Base byte addresses of this block's operand panels (advance 64B per k-step).
  const unsigned long long gaA_hi = (unsigned long long)(size_t)(Ahi  + (size_t)rowBase * K);
  const unsigned long long gaA_lo = (unsigned long long)(size_t)(Alo  + (size_t)rowBase * K);
  const unsigned long long gaB_hi = (unsigned long long)(size_t)(Bthi + (size_t)colBase * K);
  const unsigned long long gaB_lo = (unsigned long long)(size_t)(Btlo + (size_t)colBase * K);

  auto issue = [&](int buf, int kt) {
    const unsigned long long koffB = (unsigned long long)(kt * 64);  // 32 bf16 = 64 bytes
    if (wv == 0)
      tdm_issue(gaA_hi + koffB, (unsigned)(size_t)&sAhi[buf][0][0], K, M, 64);
    else if (wv == 1)
      tdm_issue(gaA_lo + koffB, (unsigned)(size_t)&sAlo[buf][0][0], K, M, 64);
    else if (wv == 2)
      tdm_issue(gaB_hi + koffB, (unsigned)(size_t)&sBhi[buf][0][0], K, N, 128);
    else if (wv == 3)
      tdm_issue(gaB_lo + koffB, (unsigned)(size_t)&sBlo[buf][0][0], K, N, 128);
  };
  issue(0, 0);
  for (int kt = 0; kt < KT; ++kt) {
    const int buf = kt & 1;
    __builtin_amdgcn_s_wait_tensorcnt(0);   // owner's DMA for `buf` complete
    __syncthreads();                        // publish buf; all readers done with buf^1
    if (kt + 1 < KT) issue(buf ^ 1, kt + 1); // overlap next DMA with compute
    compute_tile(buf);
  }
#else
  // Fallback: packed b128 copy staging (single buffer)
  for (int kt = 0; kt < KT; ++kt) {
    const int k0 = kt * 32;
    {
      const int r = tid >> 2, kq = (tid & 3) << 3;
      *(v8bf*)&sAhi[0][r][kq] = *(const v8bf*)(Ahi + (size_t)(rowBase + r) * K + k0 + kq);
      *(v8bf*)&sAlo[0][r][kq] = *(const v8bf*)(Alo + (size_t)(rowBase + r) * K + k0 + kq);
    }
    #pragma unroll
    for (int c = 0; c < 2; ++c) {
      const int idx = tid + (c << 8);
      const int rn = idx >> 2, kq = (idx & 3) << 3;
      *(v8bf*)&sBhi[0][rn][kq] = *(const v8bf*)(Bthi + (size_t)(colBase + rn) * K + k0 + kq);
      *(v8bf*)&sBlo[0][rn][kq] = *(const v8bf*)(Btlo + (size_t)(colBase + rn) * K + k0 + kq);
    }
    __syncthreads();
    compute_tile(0);
    __syncthreads();
  }
#endif

  // ---- epilogue: C/D layout: VGPR v -> M = v + 8*hl, N = ln ----
  #pragma unroll
  for (int mi = 0; mi < 2; ++mi) {
    #pragma unroll
    for (int v = 0; v < 8; ++v) {
      const int row = rowBase + waveM * 32 + mi * 16 + v + 8 * hl;
      if (EPI == 2) {
        float s = acc[mi][0][v] * acc[mi][0][v] + acc[mi][1][v] * acc[mi][1][v];
        s += __shfl_xor(s, 1, 32);
        s += __shfl_xor(s, 2, 32);
        s += __shfl_xor(s, 4, 32);
        s += __shfl_xor(s, 8, 32);
        if (ln == 0) atomicAdd(&Vout[row], s);
      } else {
        #pragma unroll
        for (int ni = 0; ni < 2; ++ni) {
          const int col = colBase + waveN * 32 + ni * 16 + ln;
          const size_t o = (size_t)row * N + col;
          float c = acc[mi][ni][v] + bias[col];
          if (EPI == 0) {
            float t = tanhf(c);
            __bf16 h = (__bf16)t;
            outHi[o] = h;
            outLo[o] = (__bf16)(t - (float)h);
          } else {
            float f = resid[o] + c;
            outF[o] = f;
            __bf16 h = (__bf16)f;
            outHi[o] = h;
            outLo[o] = (__bf16)(f - (float)h);
          }
        }
      }
    }
  }
}

// ---------------------------------------------------------------------------
__global__ void zero_kernel(float* p, int n) {
  int i = blockIdx.x * blockDim.x + threadIdx.x;
  if (i < n) p[i] = 0.0f;
}

__device__ __forceinline__ float sgnf(float x) {
  return (x > 0.0f) ? 1.0f : ((x < 0.0f) ? -1.0f : 0.0f);
}

// Exact emulation of the reference masked Newton+bisection loop, using
// V(fhatx*a) == a^2*Vf and dV/da == 2*a*Vf (exact for the quadratic form).
__global__ void alpha_kernel(const float* __restrict__ Vx, const float* __restrict__ Vf,
                             float* __restrict__ alpha, int n)
{
  int i = blockIdx.x * blockDim.x + threadIdx.x;
  if (i >= n) return;
  const float vf = Vf[i];
  const float t  = BETA_C * Vx[i];
  float a = 1.0f, e1 = 0.0f, e2 = 1.0f;
  bool m = (vf - t) > 0.0f;
  #pragma unroll 1
  for (int it = 0; it < MAXIT; ++it) {
    if (m) a = a - (a * a * vf - t) / (2.0f * a * vf);   // masked Newton
    if (a < e1 || a > e2) {                              // bisection fallback
      const float ab = e1 + (e2 - e1) * 0.5f;
      const float sa = sgnf(ab * ab * vf - t);
      const float s1 = sgnf(e1 * e1 * vf - t);
      const float s2 = sgnf(e2 * e2 * vf - t);
      if (sa * s1 < 0.0f) e2 = ab;
      if (sa * s2 < 0.0f) e1 = ab;
      a = ab;
    }
    m = fabsf(a * a * vf - t) > TOL_C;                   // re-mask
  }
  alpha[i] = a;
}

__global__ void scale_kernel(const float* __restrict__ f, const float* __restrict__ alpha,
                             float* __restrict__ out, int vec_per_row)
{
  int idx = blockIdx.x * blockDim.x + threadIdx.x;
  int row = idx / vec_per_row;
  const float a = alpha[row];
  float4 v = ((const float4*)f)[idx];
  v.x *= a; v.y *= a; v.z *= a; v.w *= a;
  ((float4*)out)[idx] = v;
}

// ---------------------------------------------------------------------------
extern "C" void kernel_launch(void* const* d_in, const int* in_sizes, int n_in,
                              void* d_out, int out_size, void* d_ws, size_t ws_size,
                              hipStream_t stream)
{
  const float* x  = (const float*)d_in[0];   // (8192, 1, 256)
  const float* W1 = (const float*)d_in[1];   // (256, 1024)
  const float* b1 = (const float*)d_in[2];   // (1024)
  const float* W2 = (const float*)d_in[3];   // (1024, 256)
  const float* b2 = (const float*)d_in[4];   // (256)
  const float* Wv = (const float*)d_in[5];   // (256, 256)
  float* out = (float*)d_out;

  // Workspace carve (256B aligned), ~61 MB total
  char* p = (char*)d_ws;
  auto carve = [&](size_t bytes) -> void* {
    void* r = (void*)p; p += (bytes + 255) & ~(size_t)255; return r;
  };
  __bf16* xhi   = (__bf16*)carve((size_t)BB * NND * 2);
  __bf16* xlo   = (__bf16*)carve((size_t)BB * NND * 2);
  __bf16* w1thi = (__bf16*)carve((size_t)HHD * NND * 2);
  __bf16* w1tlo = (__bf16*)carve((size_t)HHD * NND * 2);
  __bf16* w2thi = (__bf16*)carve((size_t)NND * HHD * 2);
  __bf16* w2tlo = (__bf16*)carve((size_t)NND * HHD * 2);
  __bf16* wvthi = (__bf16*)carve((size_t)NND * NND * 2);
  __bf16* wvtlo = (__bf16*)carve((size_t)NND * NND * 2);
  __bf16* hhi   = (__bf16*)carve((size_t)BB * HHD * 2);
  __bf16* hlo   = (__bf16*)carve((size_t)BB * HHD * 2);
  float*  fhat  = (float*)carve((size_t)BB * NND * 4);
  __bf16* fhhi  = (__bf16*)carve((size_t)BB * NND * 2);
  __bf16* fhlo  = (__bf16*)carve((size_t)BB * NND * 2);
  float*  Vx    = (float*)carve((size_t)BB * 4);
  float*  Vf    = (float*)carve((size_t)BB * 4);
  float*  alpha = (float*)carve((size_t)BB * 4);

  // --- prep: split x; transpose+split weights (WMMA B wants N-major) ---
  split_kernel<<<(BB * NND) / 256, 256, 0, stream>>>(x, xhi, xlo, BB * NND);
  transpose_split_kernel<<<dim3(HHD / 32, NND / 32), 256, 0, stream>>>(W1, w1thi, w1tlo, NND, HHD);
  transpose_split_kernel<<<dim3(NND / 32, HHD / 32), 256, 0, stream>>>(W2, w2thi, w2tlo, HHD, NND);
  transpose_split_kernel<<<dim3(NND / 32, NND / 32), 256, 0, stream>>>(Wv, wvthi, wvtlo, NND, NND);
  zero_kernel<<<(2 * BB) / 256, 256, 0, stream>>>(Vx, 2 * BB);

  // h = tanh(x @ W1 + b1)        (split bf16 out)
  gemm_bf16x3<0><<<dim3(HHD / 128, BB / 64), 256, 0, stream>>>(
      xhi, xlo, w1thi, w1tlo, b1, nullptr, nullptr, hhi, hlo, nullptr, BB, HHD, NND);
  // fhatx = x + h @ W2 + b2      (fp32 + split bf16 out)
  gemm_bf16x3<1><<<dim3(NND / 128, BB / 64), 256, 0, stream>>>(
      hhi, hlo, w2thi, w2tlo, b2, x, fhat, fhhi, fhlo, nullptr, BB, NND, HHD);
  // Vx[i] = ||x_i @ Wv||^2
  gemm_bf16x3<2><<<dim3(NND / 128, BB / 64), 256, 0, stream>>>(
      xhi, xlo, wvthi, wvtlo, nullptr, nullptr, nullptr, nullptr, nullptr, Vx, BB, NND, NND);
  // Vf[i] = ||fhatx_i @ Wv||^2
  gemm_bf16x3<2><<<dim3(NND / 128, BB / 64), 256, 0, stream>>>(
      fhhi, fhlo, wvthi, wvtlo, nullptr, nullptr, nullptr, nullptr, nullptr, Vf, BB, NND, NND);

  // Newton/bisection per sample, then out = fhatx * alpha
  alpha_kernel<<<BB / 256, 256, 0, stream>>>(Vx, Vf, alpha, BB);
  scale_kernel<<<(BB * (NND / 4)) / 256, 256, 0, stream>>>(fhat, alpha, out, NND / 4);
}
